// SGFormer_21646635172730
// MI455X (gfx1250) — compile-verified, matrix operand order checked by reading, not run on hardware
//
#include <hip/hip_runtime.h>

typedef unsigned short u16;
typedef __attribute__((ext_vector_type(8)))  u16    v8u;
typedef __attribute__((ext_vector_type(16))) u16    v16u;
typedef __attribute__((ext_vector_type(16))) __bf16 v16bf;
typedef __attribute__((ext_vector_type(8)))  float  v8f;

__device__ __forceinline__ u16 f2bf(float f) {
  unsigned u = __float_as_uint(f);
  u += 0x7FFFu + ((u >> 16) & 1u);          // round-to-nearest-even
  return (u16)(u >> 16);
}
__device__ __forceinline__ float bf2f(u16 h) {
  return __uint_as_float(((unsigned)h) << 16);
}

__device__ __forceinline__ float block_reduce_256(float v, float* sh) {
  int t = threadIdx.x;
  sh[t] = v; __syncthreads();
#pragma unroll
  for (int s = 128; s > 0; s >>= 1) {
    if (t < s) sh[t] += sh[t + s];
    __syncthreads();
  }
  float r = sh[0];
  __syncthreads();
  return r;
}

// Build a 16x32 (or 32x16) bf16 WMMA fragment from an LDS row.
// Per ISA: lanes 0-15 hold K={0..7,16..23}, lanes 16-31 hold K={8..15,24..31}.
__device__ __forceinline__ v16bf frag_lds(const u16* rowp, int klo) {
  v8u a = *(const v8u*)(rowp + klo);
  v8u b = *(const v8u*)(rowp + klo + 16);
  union { v16u u; v16bf f; } cv;
#pragma unroll
  for (int i = 0; i < 8; ++i) { cv.u[i] = a[i]; cv.u[i + 8] = b[i]; }
  return cv.f;
}

#define LDP 40  // LDS row stride (bf16 elems): 80B rows, 16B aligned, conflict-friendly

// C[M,N](f32) = A[M,K](bf16) @ B[K,N](bf16)  (+bias), optional bf16 mirror of C.
// TRANS_A: logical A[m,l] = Asrc[l*lda + m]  (transpose through LDS).
// ATOMIC_C: split-K, atomicAdd f32 partials into pre-zeroed C.
template <bool TRANS_A, bool ATOMIC_C>
__global__ __launch_bounds__(256)
void sg_gemm(const u16* __restrict__ A, long lda, long sA,
             const u16* __restrict__ B, long ldb, long sB,
             float* __restrict__ C, long ldc, long sC,
             u16* __restrict__ Cb, long ldcb, long sCb,
             const float* __restrict__ bias,
             int M, int N, int K, int chunkK) {
  __shared__ __align__(16) u16 As[128 * LDP];
  __shared__ __align__(16) u16 Bs[128 * LDP];

  const int MT = (M + 127) >> 7;
  const int tileM = blockIdx.x % MT, tileN = blockIdx.x / MT;
  const int m0 = tileM << 7, n0 = tileN << 7;
  const long z = blockIdx.z;
  A += z * sA; B += z * sB;
  const int k0 = blockIdx.y * chunkK;
  const int kend = (k0 + chunkK < K) ? (k0 + chunkK) : K;

  const int tid = threadIdx.x;
  const int lane = tid & 31, wave = tid >> 5;
  const int wm = (wave & 3) << 5;   // 4 waves along M: 0,32,64,96
  const int wn = (wave >> 2) << 6;  // 2 waves along N: 0,64

  const bool aFullM = (m0 + 128 <= M) && ((lda & 7) == 0);
  const bool bFullN = (n0 + 128 <= N) && ((ldb & 7) == 0);

  v8f acc[2][4];
#pragma unroll
  for (int a = 0; a < 2; ++a)
#pragma unroll
    for (int b = 0; b < 4; ++b)
#pragma unroll
      for (int r = 0; r < 8; ++r) acc[a][b][r] = 0.f;

  for (int kk = k0; kk < kend; kk += 32) {
    const bool fullK = (kk + 32 <= kend);
    // ---- A tile ----
    if (!TRANS_A) {
      if (aFullM && fullK) {
        // 128x32 tile: 512 chunks of 8 bf16 (16B), 2 per thread, b128 loads
#pragma unroll
        for (int i = 0; i < 2; ++i) {
          int c = tid + i * 256;
          int row = c >> 2, col = (c & 3) << 3;
          v8u v = *(const v8u*)(A + (long)(m0 + row) * lda + kk + col);
          *(v8u*)&As[row * LDP + col] = v;
        }
      } else {
#pragma unroll
        for (int i = 0; i < 16; ++i) {
          int e = tid + i * 256;
          int col = e & 31, row = e >> 5;
          int gr = m0 + row, gc = kk + col;
          u16 v = (gr < M && gc < kend) ? A[(long)gr * lda + gc] : (u16)0;
          As[row * LDP + col] = v;
        }
      }
    } else {
      if (aFullM && fullK) {
        // read 32 source rows x 128 cols as b128, scatter-transpose into LDS
#pragma unroll
        for (int i = 0; i < 2; ++i) {
          int c = tid + i * 256;
          int rl = c >> 4, col = (c & 15) << 3;
          v8u v = *(const v8u*)(A + (long)(kk + rl) * lda + m0 + col);
#pragma unroll
          for (int j = 0; j < 8; ++j) As[(col + j) * LDP + rl] = v[j];
        }
      } else {
#pragma unroll
        for (int i = 0; i < 16; ++i) {
          int e = tid + i * 256;
          int cm = e & 127, rl = e >> 7;
          int gr = kk + rl, gc = m0 + cm;
          u16 v = (gr < kend && gc < M) ? A[(long)gr * lda + gc] : (u16)0;
          As[cm * LDP + rl] = v;
        }
      }
    }
    // ---- B tile (transposed into LDS: Bs[n][k]) ----
    if (bFullN && fullK) {
#pragma unroll
      for (int i = 0; i < 2; ++i) {
        int c = tid + i * 256;
        int rk = c >> 4, col = (c & 15) << 3;
        v8u v = *(const v8u*)(B + (long)(kk + rk) * ldb + n0 + col);
#pragma unroll
        for (int j = 0; j < 8; ++j) Bs[(col + j) * LDP + rk] = v[j];
      }
    } else {
#pragma unroll
      for (int i = 0; i < 16; ++i) {
        int e = tid + i * 256;
        int cn = e & 127, rk = e >> 7;
        int gr = kk + rk, gc = n0 + cn;
        u16 v = (gr < kend && gc < N) ? B[(long)gr * ldb + gc] : (u16)0;
        Bs[cn * LDP + rk] = v;
      }
    }
    __syncthreads();

    const int rsel = lane & 15;
    const int klo = (lane < 16) ? 0 : 8;
    v16bf af[2], bfm[4];
#pragma unroll
    for (int a = 0; a < 2; ++a) af[a] = frag_lds(&As[(wm + a * 16 + rsel) * LDP], klo);
#pragma unroll
    for (int b = 0; b < 4; ++b) bfm[b] = frag_lds(&Bs[(wn + b * 16 + rsel) * LDP], klo);
#pragma unroll
    for (int a = 0; a < 2; ++a)
#pragma unroll
      for (int b = 0; b < 4; ++b)
        acc[a][b] = __builtin_amdgcn_wmma_f32_16x16x32_bf16(
            false, af[a], false, bfm[b], (short)0, acc[a][b], false, false);
    __syncthreads();
  }

  // C/D layout: VGPR r -> M = r (lanes 0-15) / r+8 (lanes 16-31), N = lane%16
  const int mbase = (lane < 16) ? 0 : 8;
  const int cn = lane & 15;
  float* Cz = C ? (C + z * sC) : nullptr;
  u16* Cbz = Cb ? (Cb + z * sCb) : nullptr;
  const bool fullC = (m0 + 128 <= M) && (n0 + 128 <= N);
#pragma unroll
  for (int a = 0; a < 2; ++a)
#pragma unroll
    for (int b = 0; b < 4; ++b)
#pragma unroll
      for (int r = 0; r < 8; ++r) {
        int gm = m0 + wm + a * 16 + mbase + r;
        int gn = n0 + wn + b * 16 + cn;
        if (fullC || (gm < M && gn < N)) {
          float v = acc[a][b][r];
          if (ATOMIC_C) {
            atomicAdd(&Cz[(long)gm * ldc + gn], v);
          } else {
            if (bias) v += bias[gn];
            if (Cz)  Cz[(long)gm * ldc + gn] = v;
            if (Cbz) Cbz[(long)gm * ldcb + gn] = f2bf(v);
          }
        }
      }
}

// ---------- elementwise / reduction kernels ----------

__global__ void sg_cvt_f2bf(const float* __restrict__ X, u16* __restrict__ Y, long n) {
  long i = (long)blockIdx.x * blockDim.x + threadIdx.x;
  long stride = (long)gridDim.x * blockDim.x;
  for (; i < n; i += stride) Y[i] = f2bf(X[i]);
}

// convert [R,C] f32 -> [R,CP] bf16, zero-padding columns C..CP-1
__global__ void sg_cvt_pad(const float* __restrict__ X, u16* __restrict__ Y,
                           long R, int C, int CP) {
  long total = R * CP;
  long i = (long)blockIdx.x * blockDim.x + threadIdx.x;
  long stride = (long)gridDim.x * blockDim.x;
  for (; i < total; i += stride) {
    long r = i / CP;
    int c = (int)(i - r * CP);
    Y[i] = (c < C) ? f2bf(X[r * C + c]) : (u16)0;
  }
}

__global__ void sg_fill(float* __restrict__ p, long n, float v) {
  long i = (long)blockIdx.x * blockDim.x + threadIdx.x;
  long stride = (long)gridDim.x * blockDim.x;
  for (; i < n; i += stride) p[i] = v;
}

__global__ __launch_bounds__(256)
void sg_sumsq_bf16(const u16* __restrict__ X, long n, float* __restrict__ out) {
  __shared__ float sh[256];
  long i = (long)blockIdx.x * 256 + threadIdx.x;
  long stride = (long)gridDim.x * 256;
  float a = 0.f;
  for (; i < n; i += stride) { float v = bf2f(X[i]); a += v * v; }
  float r = block_reduce_256(a, sh);
  if (threadIdx.x == 0) atomicAdd(out, r);
}

// column sums of K [N,1024] (bf16) -> out[1024]; thread t owns cols t,t+256,t+512,t+768
__global__ __launch_bounds__(256)
void sg_colsum(const u16* __restrict__ K, int N, float* __restrict__ out) {
  int t = threadIdx.x;
  int r0 = blockIdx.x * 64;
  int r1 = (r0 + 64 < N) ? r0 + 64 : N;
  float a0 = 0, a1 = 0, a2 = 0, a3 = 0;
  for (int r = r0; r < r1; ++r) {
    const u16* row = K + (long)r * 1024;
    a0 += bf2f(row[t]); a1 += bf2f(row[t + 256]);
    a2 += bf2f(row[t + 512]); a3 += bf2f(row[t + 768]);
  }
  atomicAdd(&out[t], a0);       atomicAdd(&out[t + 256], a1);
  atomicAdd(&out[t + 512], a2); atomicAdd(&out[t + 768], a3);
}

// LN(g,b) + relu over [N,256]; writes f32 + bf16
__global__ __launch_bounds__(256)
void sg_ln_relu(const float* __restrict__ X, const float* __restrict__ g,
                const float* __restrict__ b, float* __restrict__ out,
                u16* __restrict__ outb) {
  __shared__ float sh[256];
  long n = blockIdx.x; int t = threadIdx.x;
  float v = X[n * 256 + t];
  float mu = block_reduce_256(v, sh) * (1.f / 256.f);
  float d = v - mu;
  float var = block_reduce_256(d * d, sh) * (1.f / 256.f);
  float y = d * rsqrtf(var + 1e-5f) * g[t] + b[t];
  y = y > 0.f ? y : 0.f;
  out[n * 256 + t] = y;
  outb[n * 256 + t] = f2bf(y);
}

// attention epilogue: h = mean_h[(s*numpart + N*V)/(s*(Q.ks_sum)+N)]; x = LN(0.5h+0.5x)
__global__ __launch_bounds__(256)
void sg_attn_epilogue(const float* __restrict__ numpart, const u16* __restrict__ Qb,
                      const float* __restrict__ Vf, const float* __restrict__ kssum,
                      const float* __restrict__ scal, const float* __restrict__ lg,
                      const float* __restrict__ lb, float* __restrict__ xcur,
                      u16* __restrict__ xcurb, float Nf) {
  __shared__ float sh[256];
  long n = blockIdx.x; int t = threadIdx.x;
  float s = rsqrtf(scal[0]) * rsqrtf(scal[1]);
  float den[4];
#pragma unroll
  for (int h = 0; h < 4; ++h) {
    long idx = n * 1024 + h * 256 + t;
    float q = bf2f(Qb[idx]);
    float d = block_reduce_256(q * kssum[h * 256 + t], sh);
    den[h] = s * d + Nf;
  }
  float hv = 0.f;
#pragma unroll
  for (int h = 0; h < 4; ++h) {
    long idx = n * 1024 + h * 256 + t;
    hv += (s * numpart[idx] + Nf * Vf[idx]) / den[h];
  }
  hv *= 0.25f;                                  // mean over heads
  float y = 0.5f * hv + 0.5f * xcur[n * 256 + t];
  float mu = block_reduce_256(y, sh) * (1.f / 256.f);
  float d0 = y - mu;
  float var = block_reduce_256(d0 * d0, sh) * (1.f / 256.f);
  float o = d0 * rsqrtf(var + 1e-5f) * lg[t] + lb[t];
  xcur[n * 256 + t] = o;
  xcurb[n * 256 + t] = f2bf(o);
}

// ---------- GCN kernels (padded row stride ld, logical width C) ----------

__global__ void sg_deg(const int* __restrict__ dst, int E, float* __restrict__ deg) {
  long i = (long)blockIdx.x * blockDim.x + threadIdx.x;
  long stride = (long)gridDim.x * blockDim.x;
  for (; i < E; i += stride) atomicAdd(&deg[dst[i]], 1.0f);
}

__global__ void sg_rsqrt(const float* __restrict__ deg, float* __restrict__ dinv, int N) {
  long i = (long)blockIdx.x * blockDim.x + threadIdx.x;
  if (i < N) dinv[i] = rsqrtf(deg[i]);
}

// out[n][*] = H[n][*] * dinv[n]^2 over [N, 512] (pad cols are zeros in H)
__global__ void sg_agg_init(const float* __restrict__ H, const float* __restrict__ dinv,
                            float* __restrict__ out, long total) {
  long i = (long)blockIdx.x * blockDim.x + threadIdx.x;
  long stride = (long)gridDim.x * blockDim.x;
  for (; i < total; i += stride) {
    long n = i >> 9;                      // stride 512
    float di = dinv[n];
    out[i] = H[i] * di * di;
  }
}

__global__ __launch_bounds__(256)
void sg_agg_edges(const float* __restrict__ H, const int* __restrict__ src,
                  const int* __restrict__ dst, const float* __restrict__ dinv,
                  float* __restrict__ out, int E, int C, int ld) {
  int e = blockIdx.x;
  if (e >= E) return;
  int s = src[e], d = dst[e];
  float c = dinv[s] * dinv[d];
  const float* hr = H + (long)s * ld;
  float* orow = out + (long)d * ld;
  for (int i = threadIdx.x; i < C; i += blockDim.x)
    atomicAdd(&orow[i], hr[i] * c);
}

// h = relu((agg + b1) * bn_scale + bn_b) over logical [N, C] with row stride ld
__global__ void sg_bn_relu(const float* __restrict__ agg, const float* __restrict__ b1,
                           const float* __restrict__ bng, const float* __restrict__ bnb,
                           float* __restrict__ out, long total, int C, int ld) {
  long i = (long)blockIdx.x * blockDim.x + threadIdx.x;
  long stride = (long)gridDim.x * blockDim.x;
  float inv = rsqrtf(1.0f + 1e-5f);
  for (; i < total; i += stride) {
    long n = i / C;
    int c = (int)(i - n * C);
    long a = n * ld + c;
    float v = (agg[a] + b1[c]) * (bng[c] * inv) + bnb[c];
    out[a] = v > 0.f ? v : 0.f;
  }
}

__global__ __launch_bounds__(256)
void sg_dot(const float* __restrict__ H, const float* __restrict__ w,
            float* __restrict__ z, int C, int ld) {
  __shared__ float sh[256];
  long n = blockIdx.x;
  float a = 0.f;
  for (int c = threadIdx.x; c < C; c += 256) a += H[n * ld + c] * w[c];
  float r = block_reduce_256(a, sh);
  if (threadIdx.x == 0) z[n] = r;
}

__global__ void sg_x2_init(const float* __restrict__ z, const float* __restrict__ dinv,
                           float* __restrict__ x2, int N) {
  long i = (long)blockIdx.x * blockDim.x + threadIdx.x;
  if (i < N) { float di = dinv[i]; x2[i] = z[i] * di * di; }
}

__global__ void sg_x2_edges(const float* __restrict__ z, const int* __restrict__ src,
                            const int* __restrict__ dst, const float* __restrict__ dinv,
                            float* __restrict__ x2, int E) {
  long i = (long)blockIdx.x * blockDim.x + threadIdx.x;
  long stride = (long)gridDim.x * blockDim.x;
  for (; i < E; i += stride) {
    int s = src[i], d = dst[i];
    atomicAdd(&x2[d], z[s] * dinv[s] * dinv[d]);
  }
}

__global__ __launch_bounds__(256)
void sg_final_reduce(const float* __restrict__ x1, const float* __restrict__ x2,
                     const float* __restrict__ gb2, const float* __restrict__ Wfc,
                     float* __restrict__ accum) {
  __shared__ float sh[256];
  long n = blockIdx.x; int t = threadIdx.x;
  float x2f = x2[n] + gb2[0];
  float v = (0.8f * x2f + 0.2f * x1[n * 256 + t]) * Wfc[t];
  float r = block_reduce_256(v, sh);
  if (t == 0) atomicAdd(accum, r);
}

__global__ void sg_finalize(const float* __restrict__ accum, const float* __restrict__ bfc,
                            float* __restrict__ out, float invN) {
  out[0] = accum[0] * invN + bfc[0];
}

// ---------- host ----------

static inline int gsb(long n) {  // grid-stride block count
  long b = (n + 255) / 256;
  return (int)(b > 8192 ? 8192 : (b < 1 ? 1 : b));
}

extern "C" void kernel_launch(void* const* d_in, const int* in_sizes, int n_in,
                              void* d_out, int out_size, void* d_ws, size_t ws_size,
                              hipStream_t stream) {
  const int N = in_sizes[0] / 1024;
  const int E = in_sizes[2] / 2;
  const float Nf = (float)N;
  const int GP = 512;   // padded GCN hidden (logical 500)

  const float* x_in = (const float*)d_in[0];
  const int* edges = (const int*)d_in[2];
  const int* src = edges;
  const int* dst = edges + E;
  const float* W0 = (const float*)d_in[4];
  const float* b0 = (const float*)d_in[5];
  const float* g0 = (const float*)d_in[6];
  const float* be0 = (const float*)d_in[7];
  const float* Wq = (const float*)d_in[8];
  const float* bq = (const float*)d_in[9];
  const float* Wk = (const float*)d_in[10];
  const float* bk = (const float*)d_in[11];
  const float* Wv = (const float*)d_in[12];
  const float* bv = (const float*)d_in[13];
  const float* lg = (const float*)d_in[14];
  const float* lb = (const float*)d_in[15];
  const float* gW1 = (const float*)d_in[16];
  const float* gb1 = (const float*)d_in[17];
  const float* bng = (const float*)d_in[18];
  const float* bnb = (const float*)d_in[19];
  const float* gW2 = (const float*)d_in[20];
  const float* gb2 = (const float*)d_in[21];
  const float* Wfc = (const float*)d_in[22];
  const float* bfc = (const float*)d_in[23];
  float* out = (float*)d_out;

  // workspace bump allocator (256B aligned blocks -> b128-friendly)
  char* p = (char*)d_ws;
  auto alloc = [&](size_t bytes) -> char* {
    char* r = p;
    p += (bytes + 255) & ~(size_t)255;
    return r;
  };
  u16* xb16   = (u16*)alloc((size_t)N * 1024 * 2);
  u16* W0b    = (u16*)alloc((size_t)1024 * 256 * 2);
  u16* gW1b   = (u16*)alloc((size_t)1024 * GP * 2);  // padded [1024,512]
  u16* Wqb    = (u16*)alloc((size_t)256 * 1024 * 2);
  u16* Wkb    = (u16*)alloc((size_t)256 * 1024 * 2);
  u16* Wvb    = (u16*)alloc((size_t)256 * 1024 * 2);
  float* xcur = (float*)alloc((size_t)N * 256 * 4);
  u16* xcurb  = (u16*)alloc((size_t)N * 256 * 2);
  u16* Qb     = (u16*)alloc((size_t)N * 1024 * 2);
  u16* Kb     = (u16*)alloc((size_t)N * 1024 * 2);
  u16* Vb     = (u16*)alloc((size_t)N * 1024 * 2);
  float* Vf   = (float*)alloc((size_t)N * 1024 * 4);
  float* nump = (float*)alloc((size_t)N * 1024 * 4);
  float* kvs  = (float*)alloc((size_t)4 * 65536 * 4);
  u16* kvsb   = (u16*)alloc((size_t)4 * 65536 * 2);
  float* kssum = (float*)alloc(1024 * 4);
  float* scal  = (float*)alloc(2 * 4);
  float* deg   = (float*)alloc((size_t)N * 4);
  float* dinv  = (float*)alloc((size_t)N * 4);
  float* zv    = (float*)alloc((size_t)N * 4);
  float* x2    = (float*)alloc((size_t)N * 4);
  float* accum = (float*)alloc(4);
  // aliases: GCN runs fully before any trans-layer use of these buffers
  float* Hg   = nump;   // [N,512] padded; later reused as t0 [N,256] then numpart
  float* agg1 = Vf;     // [N,512] padded; later reused as V f32

  auto gemm_nn = [&](const u16* A, long lda, long sA, const u16* B, long ldb, long sB,
                     float* C, long ldc, long sC, u16* Cb, long ldcb, long sCb,
                     const float* bias, int M, int Nn, int K, int batch) {
    int MT = (M + 127) / 128, NT = (Nn + 127) / 128;
    dim3 g(MT * NT, 1, batch);
    sg_gemm<false, false><<<g, 256, 0, stream>>>(A, lda, sA, B, ldb, sB, C, ldc, sC,
                                                 Cb, ldcb, sCb, bias, M, Nn, K, K);
  };

  // ---------------- GCN branch ----------------
  sg_cvt_f2bf<<<gsb((long)N * 1024), 256, 0, stream>>>(x_in, xb16, (long)N * 1024);
  sg_cvt_pad<<<gsb((long)1024 * GP), 256, 0, stream>>>(gW1, gW1b, 1024, 500, GP);
  // padded GEMM: N=512 so all B/C tiles take the aligned b128 fast path
  gemm_nn(xb16, 1024, 0, gW1b, GP, 0, Hg, GP, 0, nullptr, 0, 0, nullptr, N, GP, 1024, 1);

  sg_fill<<<gsb(N), 256, 0, stream>>>(deg, N, 1.0f);
  sg_deg<<<gsb(E), 256, 0, stream>>>(dst, E, deg);
  sg_rsqrt<<<(N + 255) / 256, 256, 0, stream>>>(deg, dinv, N);

  sg_agg_init<<<gsb((long)N * GP), 256, 0, stream>>>(Hg, dinv, agg1, (long)N * GP);
  sg_agg_edges<<<E, 256, 0, stream>>>(Hg, src, dst, dinv, agg1, E, 500, GP);
  sg_bn_relu<<<gsb((long)N * 500), 256, 0, stream>>>(agg1, gb1, bng, bnb, Hg,
                                                     (long)N * 500, 500, GP);
  sg_dot<<<N, 256, 0, stream>>>(Hg, gW2, zv, 500, GP);
  sg_x2_init<<<(N + 255) / 256, 256, 0, stream>>>(zv, dinv, x2, N);
  sg_x2_edges<<<gsb(E), 256, 0, stream>>>(zv, src, dst, dinv, x2, E);

  // ---------------- trans branch ----------------
  float* t0 = nump;  // [N,256], dead before numpart is needed
  sg_cvt_f2bf<<<gsb(1024 * 256), 256, 0, stream>>>(W0, W0b, 1024 * 256);
  gemm_nn(xb16, 1024, 0, W0b, 256, 0, t0, 256, 0, nullptr, 0, 0, b0, N, 256, 1024, 1);
  sg_ln_relu<<<N, 256, 0, stream>>>(t0, g0, be0, xcur, xcurb);

  for (int i = 0; i < 2; ++i) {
    const long woff = (long)i * 256 * 1024;
    sg_cvt_f2bf<<<gsb(256 * 1024), 256, 0, stream>>>(Wq + woff, Wqb, 256 * 1024);
    sg_cvt_f2bf<<<gsb(256 * 1024), 256, 0, stream>>>(Wk + woff, Wkb, 256 * 1024);
    sg_cvt_f2bf<<<gsb(256 * 1024), 256, 0, stream>>>(Wv + woff, Wvb, 256 * 1024);

    // Q/K -> bf16 only; V -> f32 + bf16
    gemm_nn(xcurb, 256, 0, Wqb, 1024, 0, nullptr, 0, 0, Qb, 1024, 0, bq + i * 1024, N, 1024, 256, 1);
    gemm_nn(xcurb, 256, 0, Wkb, 1024, 0, nullptr, 0, 0, Kb, 1024, 0, bk + i * 1024, N, 1024, 256, 1);
    gemm_nn(xcurb, 256, 0, Wvb, 1024, 0, Vf, 1024, 0, Vb, 1024, 0, bv + i * 1024, N, 1024, 256, 1);

    sg_fill<<<1, 256, 0, stream>>>(scal, 2, 0.f);
    sg_sumsq_bf16<<<2048, 256, 0, stream>>>(Qb, (long)N * 1024, scal + 0);
    sg_sumsq_bf16<<<2048, 256, 0, stream>>>(Kb, (long)N * 1024, scal + 1);
    sg_fill<<<4, 256, 0, stream>>>(kssum, 1024, 0.f);
    sg_colsum<<<(N + 63) / 64, 256, 0, stream>>>(Kb, N, kssum);

    // kvs[h] = K_h^T @ V_h  (split-K over nodes, atomic f32 accumulate)
    sg_fill<<<gsb(4 * 65536), 256, 0, stream>>>(kvs, 4 * 65536, 0.f);
    {
      const int CK = 4096;
      dim3 g(4, (N + CK - 1) / CK, 4);
      sg_gemm<true, true><<<g, 256, 0, stream>>>(
          Kb, 1024, 256, Vb, 1024, 256, kvs, 256, 65536,
          nullptr, 0, 0, nullptr, 256, 256, N, CK);
    }
    sg_cvt_f2bf<<<gsb(4 * 65536), 256, 0, stream>>>(kvs, kvsb, 4 * 65536);

    // numpart[:,h,:] = Q_h @ kvs[h]   (batched over heads)
    {
      dim3 g(((N + 127) / 128) * 2, 1, 4);
      sg_gemm<false, false><<<g, 256, 0, stream>>>(
          Qb, 1024, 256, kvsb, 256, 65536, nump, 1024, 256,
          nullptr, 0, 0, nullptr, N, 256, 256, 256);
    }

    sg_attn_epilogue<<<N, 256, 0, stream>>>(nump, Qb, Vf, kssum, scal,
                                            lg + i * 256, lb + i * 256, xcur, xcurb, Nf);
  }

  // ---------------- combine + output ----------------
  sg_fill<<<1, 256, 0, stream>>>(accum, 1, 0.f);
  sg_final_reduce<<<N, 256, 0, stream>>>(xcur, x2, gb2, Wfc, accum);
  sg_finalize<<<1, 1, 0, stream>>>(accum, bfc, out, 1.0f / Nf);

  (void)n_in; (void)out_size; (void)ws_size;
}
